// Decoder_22299470201375
// MI455X (gfx1250) — compile-verified, matrix operand order checked by reading, not run on hardware
//
#include <hip/hip_runtime.h>
#include <hip/hip_bf16.h>

typedef _Float16 v16h __attribute__((ext_vector_type(16)));
typedef _Float16 v8h  __attribute__((ext_vector_type(8)));
typedef float    v8f  __attribute__((ext_vector_type(8)));

constexpr int B_  = 32;
constexpr int TD  = 2048;   // decoder steps
constexpr int TE  = 1024;   // encoder steps
constexpr int D_  = 256;    // model dim
constexpr int IN_ = 80;     // mel dim
constexpr int INP = 96;     // mel dim padded to 3*32 for K-loop
constexpr int D2  = 512;    // conv out channels (2*D)

#define WMMA_F16(a,b,c) __builtin_amdgcn_wmma_f32_16x16x32_f16(false,(a),false,(b),(short)0,(c),false,false)

// ---- fragment loaders: 16-bit 16x32 A/B layout, two b128 loads per lane ----
// lanes 0-15: row/col = lane, K = [k0..k0+7] and [k0+16..k0+23]
// lanes 16-31: row/col = lane-16, K = [k0+8..k0+15] and [k0+24..k0+31]
__device__ __forceinline__ v16h frag_ld(const _Float16* __restrict__ base, int stride, int lane) {
  const int r  = lane & 15;
  const int kh = (lane >> 4) << 3;
  const _Float16* p = base + (size_t)r * stride + kh;
  union { v16h v; v8h h[2]; } u;
  u.h[0] = *(const v8h*)(p);
  u.h[1] = *(const v8h*)(p + 16);
  return u.v;
}

// row-guarded variant, used ONLY in the (wave-uniform) first-time-tile path
__device__ __forceinline__ v16h frag_ld_rows(const _Float16* __restrict__ base, int stride, int lane, int row0) {
  const int r  = row0 + (lane & 15);
  const int kh = (lane >> 4) << 3;
  v16h out = {};
  if (r >= 0) {
    const _Float16* p = base + (size_t)r * stride + kh;
    union { v16h v; v8h h[2]; } u;
    u.h[0] = *(const v8h*)(p);
    u.h[1] = *(const v8h*)(p + 16);
    out = u.v;
  }
  return out;
}

// ============================ conversion kernels ============================
__global__ void k_cvt(const float* __restrict__ in, _Float16* __restrict__ out, int n) {
  int i = blockIdx.x * 256 + threadIdx.x;
  if (i < n) out[i] = (_Float16)in[i];
}

__global__ void k_wlin_pad(const float* __restrict__ w, _Float16* __restrict__ o) {
  // W_lin (256,80) -> (256,96) zero padded
  int i = blockIdx.x * 256 + threadIdx.x;
  if (i < D_ * INP) {
    int r = i / INP, c = i % INP;
    o[i] = (_Float16)((c < IN_) ? w[r * IN_ + c] : 0.0f);
  }
}

__global__ void k_conv_w_cvt(const float* __restrict__ w, _Float16* __restrict__ o) {
  // (512,256,5) [n][c][j] -> (5,512,256) [j][n][c]
  int i = blockIdx.x * 256 + threadIdx.x;
  if (i < D2 * D_ * 5) {
    int j = i % 5, c = (i / 5) % D_, n = i / (5 * D_);
    o[((size_t)j * D2 + n) * D_ + c] = (_Float16)w[i];
  }
}

__global__ void k_mel_shift(const float* __restrict__ mel, _Float16* __restrict__ o) {
  // melh[b][t][c] = mel[b][t-1][c] (t>0, c<80) else 0 ; padded to 96 cols
  int i = blockIdx.x * 256 + threadIdx.x;
  if (i >= B_ * TD * INP) return;
  int c = i % INP, t = (i / INP) % TD, b = i / (INP * TD);
  float v = 0.0f;
  if (t > 0 && c < IN_) v = mel[((size_t)b * TD + (t - 1)) * IN_ + c];
  o[i] = (_Float16)v;
}

__global__ void k_make_enc(const float* __restrict__ enc, const float* __restrict__ emb,
                           _Float16* __restrict__ ench, _Float16* __restrict__ vth) {
  int i = blockIdx.x * 256 + threadIdx.x;
  if (i >= B_ * TE * D_) return;
  int d = i % D_, s = (i / D_) % TE, b = i / (D_ * TE);
  float e = enc[i];
  ench[i] = (_Float16)e;                                       // (b, s, d) for QK^T B-frags
  vth[((size_t)b * D_ + d) * TE + s] = (_Float16)(e + emb[i]); // (b, d, s) for PV B-frags
}

__global__ void k_zero_row0(_Float16* __restrict__ xh) {
  // reference prepends zeros BEFORE the linear, so row 0 must be exactly 0 (not bias)
  int i = blockIdx.x * 256 + threadIdx.x;
  if (i < B_ * D_) xh[(size_t)(i / D_) * TD * D_ + (i % D_)] = (_Float16)0.0f;
}

// ============================ generic WMMA GEMM =============================
// out[M,N] = A[M,K] @ W[N,K]^T + bias
// one wave = 64 rows x 16 cols (4 M-tiles share each B fragment: 4 wmma / B load)
template <bool F32OUT>
__global__ void k_gemm(const _Float16* __restrict__ A, const _Float16* __restrict__ W,
                       const float* __restrict__ bias, void* __restrict__ outp,
                       int M, int N, int K) {
  int wave = blockIdx.x * 4 + (threadIdx.x >> 5);
  int lane = threadIdx.x & 31;
  int ntiles = N >> 4;
  int tm = wave / ntiles, tn = wave % ntiles;
  if (tm * 64 >= M) return;                       // wave-uniform
  const _Float16* Abase = A + (size_t)(tm * 64) * K;
  const _Float16* Wbase = W + (size_t)(tn * 16) * K;
  v8f acc[4] = {};
  for (int k0 = 0; k0 < K; k0 += 32) {
    v16h b = frag_ld(Wbase + k0, K, lane);
#pragma unroll
    for (int i = 0; i < 4; i++) {
      v16h a = frag_ld(Abase + (size_t)(i * 16) * K + k0, K, lane);
      acc[i] = WMMA_F16(a, b, acc[i]);
    }
  }
  int n = tn * 16 + (lane & 15);
  float bv = bias ? bias[n] : 0.0f;
  int rsub = (lane >> 4) << 3;
#pragma unroll
  for (int i = 0; i < 4; i++) {
    int rbase = tm * 64 + i * 16 + rsub;
    if (F32OUT) {
      float* out = (float*)outp;
#pragma unroll
      for (int j = 0; j < 8; j++) out[(size_t)(rbase + j) * N + n] = acc[i][j] + bv;
    } else {
      _Float16* out = (_Float16*)outp;
#pragma unroll
      for (int j = 0; j < 8; j++) out[(size_t)(rbase + j) * N + n] = (_Float16)(acc[i][j] + bv);
    }
  }
}

// ===================== causal Conv1d(k=5, D->2D) + GLU ======================
// one wave: 32 time rows x 16 GLU output cols. Each weight-pair (a,g) fragment
// feeds 4 wmma (2 time tiles x 2 gates). Row guard only in the tt==0 path.
__global__ void k_conv_glu(const _Float16* __restrict__ xin, const _Float16* __restrict__ wc,
                           const float* __restrict__ bias, _Float16* __restrict__ out) {
  int wave = blockIdx.x * 4 + (threadIdx.x >> 5);
  int lane = threadIdx.x & 31;
  int ntile = wave & 15;           // 16 tiles over D
  int tt = (wave >> 4) & 63;       // 64 tiles of 32 rows over TD
  int b = wave >> 10;              // 32 batches
  const _Float16* xb = xin + (size_t)b * TD * D_;
  int t0 = tt * 32, na = ntile * 16;
  v8f aa0 = {}, ag0 = {}, aa1 = {}, ag1 = {};
  if (tt != 0) {
    // fast path: all rows in range, branch-free fragment loads
#pragma unroll
    for (int j = 0; j < 5; j++) {
      int row0 = t0 - 4 + j;
      const _Float16* wj = wc + (size_t)j * D2 * D_;
      for (int c0 = 0; c0 < D_; c0 += 32) {
        v16h wa = frag_ld(wj + (size_t)na * D_ + c0, D_, lane);
        v16h wg = frag_ld(wj + (size_t)(na + D_) * D_ + c0, D_, lane);
        v16h x0 = frag_ld(xb + (size_t)row0 * D_ + c0, D_, lane);
        v16h x1 = frag_ld(xb + (size_t)(row0 + 16) * D_ + c0, D_, lane);
        aa0 = WMMA_F16(x0, wa, aa0);
        ag0 = WMMA_F16(x0, wg, ag0);
        aa1 = WMMA_F16(x1, wa, aa1);
        ag1 = WMMA_F16(x1, wg, ag1);
      }
    }
  } else {
    // first tile: rows t-4+j may be negative for the lower 16-row subtile
#pragma unroll
    for (int j = 0; j < 5; j++) {
      int row0 = t0 - 4 + j;
      const _Float16* wj = wc + (size_t)j * D2 * D_;
      for (int c0 = 0; c0 < D_; c0 += 32) {
        v16h wa = frag_ld(wj + (size_t)na * D_ + c0, D_, lane);
        v16h wg = frag_ld(wj + (size_t)(na + D_) * D_ + c0, D_, lane);
        v16h x0 = frag_ld_rows(xb + c0, D_, lane, row0);
        v16h x1 = frag_ld(xb + (size_t)(row0 + 16) * D_ + c0, D_, lane);
        aa0 = WMMA_F16(x0, wa, aa0);
        ag0 = WMMA_F16(x0, wg, ag0);
        aa1 = WMMA_F16(x1, wa, aa1);
        ag1 = WMMA_F16(x1, wg, ag1);
      }
    }
  }
  int n = na + (lane & 15);
  float ba = bias[n], bg = bias[n + D_];
  int rsub = (lane >> 4) << 3;
  _Float16* ob = out + (size_t)b * TD * D_;
#pragma unroll
  for (int half = 0; half < 2; half++) {
    const v8f& va = half ? aa1 : aa0;
    const v8f& vg = half ? ag1 : ag0;
    int rbase = t0 + half * 16 + rsub;
#pragma unroll
    for (int j = 0; j < 8; j++) {
      int r = rbase + j;
      float av = va[j] + ba;
      float gv = vg[j] + bg;
      float res = (float)xb[(size_t)r * D_ + n];
      float sig = 1.0f / (1.0f + __expf(-gv));
      ob[(size_t)r * D_ + n] = (_Float16)((av * sig + res) * 0.70710678118f);
    }
  }
}

// ===================== fused attention (per 16-row tile) ====================
// block = 4 waves, 16 decoder rows. S(16x1024,f32) in LDS -> softmax -> P(f16)
// in LDS -> O = P @ V via WMMA with ds-sourced A fragments. Epilogue adds h0.
constexpr int SSTR = TE + 4;    // f32 stride, 16B-aligned rows
constexpr int PSTR = TE + 16;   // f16 stride, 16B-aligned rows

__global__ void k_attn(const _Float16* __restrict__ qh, const _Float16* __restrict__ ench,
                       const _Float16* __restrict__ vth, const _Float16* __restrict__ h0h,
                       _Float16* __restrict__ hch) {
  extern __shared__ char smem_raw[];
  float*    S = (float*)smem_raw;                                  // 16*SSTR*4
  _Float16* P = (_Float16*)(smem_raw + 16 * SSTR * 4);             // 16*PSTR*2
  float*  red = (float*)(smem_raw + 16 * SSTR * 4 + 16 * PSTR * 2);// 128
  float* rowm = red + 128;                                         // 16
  float* rowl = rowm + 16;                                         // 16

  int tid = threadIdx.x;
  int w = tid >> 5, lane = tid & 31;
  int tt = blockIdx.x & 127, b = blockIdx.x >> 7;
  int t0 = tt * 16;

  const _Float16* qb = qh + (size_t)b * TD * D_ + (size_t)t0 * D_;
  const _Float16* eb = ench + (size_t)b * TE * D_;

  // resident Q fragments (16 rows x 256 K)
  v16h qf[8];
#pragma unroll
  for (int i = 0; i < 8; i++) qf[i] = frag_ld(qb + i * 32, D_, lane);

  // ---- pass A: S = Q @ K^T, 16-col chunks round-robined over 4 waves ----
  for (int ch = w; ch < TE / 16; ch += 4) {
    int s0 = ch * 16;
    v8f acc = {};
#pragma unroll
    for (int i = 0; i < 8; i++) {
      v16h bk = frag_ld(eb + (size_t)s0 * D_ + i * 32, D_, lane);
      acc = WMMA_F16(qf[i], bk, acc);
    }
    int col = s0 + (lane & 15);
    int rb = (lane >> 4) << 3;
#pragma unroll
    for (int j = 0; j < 8; j++) S[(rb + j) * SSTR + col] = acc[j];
  }
  __syncthreads();

  // ---- softmax over T_enc (128 threads: 16 rows x 8 segments of 128) ----
  int row = tid & 15, seg = tid >> 4;
  {
    float m = -1e30f;
    for (int c = seg * 128; c < seg * 128 + 128; c++) m = fmaxf(m, S[row * SSTR + c]);
    red[seg * 16 + row] = m;
  }
  __syncthreads();
  if (tid < 16) {
    float m = red[tid];
    for (int s2 = 1; s2 < 8; s2++) m = fmaxf(m, red[s2 * 16 + tid]);
    rowm[tid] = m;
  }
  __syncthreads();
  {
    float m = rowm[row], ssum = 0.0f;
    for (int c = seg * 128; c < seg * 128 + 128; c++) {
      float e = __expf(S[row * SSTR + c] - m);
      S[row * SSTR + c] = e;
      ssum += e;
    }
    red[seg * 16 + row] = ssum;
  }
  __syncthreads();
  if (tid < 16) {
    float l = red[tid];
    for (int s2 = 1; s2 < 8; s2++) l += red[s2 * 16 + tid];
    rowl[tid] = 1.0f / l;
  }
  __syncthreads();
  {
    float inv = rowl[row];
    for (int c = seg * 128; c < seg * 128 + 128; c++)
      P[row * PSTR + c] = (_Float16)(S[row * SSTR + c] * inv);
  }
  __syncthreads();

  // ---- pass B: O = P @ V ; wave w owns 64 of the 256 D columns ----
  const _Float16* vb = vth + (size_t)b * D_ * TE;
  int n0 = w * 64;
  v8f acc[4] = {};
  for (int k0 = 0; k0 < TE; k0 += 32) {
    v16h aP = frag_ld(P + k0, PSTR, lane);   // ds_load_b128 fragments
#pragma unroll
    for (int i = 0; i < 4; i++) {
      v16h bV = frag_ld(vb + (size_t)(n0 + i * 16) * TE + k0, TE, lane);
      acc[i] = WMMA_F16(aP, bV, acc[i]);
    }
  }
  const _Float16* hb = h0h + (size_t)b * TD * D_ + (size_t)t0 * D_;
  _Float16* ob = hch + (size_t)b * TD * D_ + (size_t)t0 * D_;
  int rb = (lane >> 4) << 3;
#pragma unroll
  for (int i = 0; i < 4; i++) {
    int n = n0 + i * 16 + (lane & 15);
#pragma unroll
    for (int j = 0; j < 8; j++) {
      int r = rb + j;
      float v = acc[i][j] + (float)hb[(size_t)r * D_ + n];
      ob[(size_t)r * D_ + n] = (_Float16)v;
    }
  }
}

// ================================ launcher ==================================
extern "C" void kernel_launch(void* const* d_in, const int* in_sizes, int n_in,
                              void* d_out, int out_size, void* d_ws, size_t ws_size,
                              hipStream_t stream) {
  const float* enc   = (const float*)d_in[0];
  const float* emb   = (const float*)d_in[1];
  const float* mel   = (const float*)d_in[2];
  const float* Wlin  = (const float*)d_in[3];
  const float* blin  = (const float*)d_in[4];
  const float* cw0   = (const float*)d_in[5];
  const float* cb0   = (const float*)d_in[6];
  const float* cw1   = (const float*)d_in[7];
  const float* cb1   = (const float*)d_in[8];
  const float* Wattn = (const float*)d_in[9];
  const float* battn = (const float*)d_in[10];
  const float* Wproj = (const float*)d_in[11];
  const float* bproj = (const float*)d_in[12];
  float* out = (float*)d_out;

  // workspace layout (256B aligned)
  char* ws = (char*)d_ws;
  size_t off = 0;
  auto take = [&](size_t bytes) { size_t o = off; off += (bytes + 255) & ~(size_t)255; return o; };
  _Float16* melh  = (_Float16*)(ws + take((size_t)B_ * TD * INP * 2));
  _Float16* xh    = (_Float16*)(ws + take((size_t)B_ * TD * D_ * 2));
  _Float16* h0h   = (_Float16*)(ws + take((size_t)B_ * TD * D_ * 2));
  _Float16* qh    = (_Float16*)(ws + take((size_t)B_ * TD * D_ * 2));
  _Float16* ench  = (_Float16*)(ws + take((size_t)B_ * TE * D_ * 2));
  _Float16* vth   = (_Float16*)(ws + take((size_t)B_ * TE * D_ * 2));
  _Float16* hch   = (_Float16*)(ws + take((size_t)B_ * TD * D_ * 2));
  _Float16* h1h   = (_Float16*)(ws + take((size_t)B_ * TD * D_ * 2));
  _Float16* wlinh = (_Float16*)(ws + take((size_t)D_ * INP * 2));
  _Float16* wattnh= (_Float16*)(ws + take((size_t)D_ * D_ * 2));
  _Float16* wprojh= (_Float16*)(ws + take((size_t)IN_ * D_ * 2));
  _Float16* wc0h  = (_Float16*)(ws + take((size_t)5 * D2 * D_ * 2));
  _Float16* wc1h  = (_Float16*)(ws + take((size_t)5 * D2 * D_ * 2));
  (void)ws_size; (void)in_sizes; (void)n_in; (void)out_size;

  // --- conversions / layout prep ---
  k_wlin_pad  <<<(D_ * INP + 255) / 256, 256, 0, stream>>>(Wlin, wlinh);
  k_cvt       <<<(D_ * D_ + 255) / 256, 256, 0, stream>>>(Wattn, wattnh, D_ * D_);
  k_cvt       <<<(IN_ * D_ + 255) / 256, 256, 0, stream>>>(Wproj, wprojh, IN_ * D_);
  k_conv_w_cvt<<<(D2 * D_ * 5 + 255) / 256, 256, 0, stream>>>(cw0, wc0h);
  k_conv_w_cvt<<<(D2 * D_ * 5 + 255) / 256, 256, 0, stream>>>(cw1, wc1h);
  k_mel_shift <<<(B_ * TD * INP + 255) / 256, 256, 0, stream>>>(mel, melh);
  k_make_enc  <<<(B_ * TE * D_ + 255) / 256, 256, 0, stream>>>(enc, emb, ench, vth);

  const int M = B_ * TD;                 // 65536 rows
  // x = Linear(mel shifted)             M x 96 -> M x 256  (64 rows/wave)
  k_gemm<false><<<(M / 64) * (D_ / 16) / 4, 128, 0, stream>>>(melh, wlinh, blin, xh, M, D_, INP);
  k_zero_row0<<<(B_ * D_ + 255) / 256, 256, 0, stream>>>(xh);   // x[:,0,:] = 0 exactly
  // conv layer 0 (GLU + residual): 32768 waves
  k_conv_glu<<<B_ * (TD / 32) * (D_ / 16) / 4, 128, 0, stream>>>(xh, wc0h, cb0, h0h);
  // q = Linear(h0)
  k_gemm<false><<<(M / 64) * (D_ / 16) / 4, 128, 0, stream>>>(h0h, wattnh, battn, qh, M, D_, D_);
  // attention + residual (h = h0 + c)
  size_t smem = (size_t)16 * SSTR * 4 + (size_t)16 * PSTR * 2 + (128 + 32) * sizeof(float);
  k_attn<<<B_ * (TD / 16), 128, smem, stream>>>(qh, ench, vth, h0h, hch);
  // conv layer 1
  k_conv_glu<<<B_ * (TD / 32) * (D_ / 16) / 4, 128, 0, stream>>>(hch, wc1h, cb1, h1h);
  // out = Linear(h1)  M x 256 -> M x 80 (f32)
  k_gemm<true><<<(M / 64) * (IN_ / 16) / 4, 128, 0, stream>>>(h1h, wprojh, bproj, (void*)out, M, IN_, D_);
}
